// KernalMetricLogits_77369540870222
// MI455X (gfx1250) — matrix operand order repaired and checked by reading, not compiled, not run on hardware
//
#include <hip/hip_runtime.h>

typedef __attribute__((ext_vector_type(16))) __bf16 v16bf;
typedef __attribute__((ext_vector_type(8)))  __bf16 v8bf;
typedef __attribute__((ext_vector_type(8)))  float  v8f;

#define B_ROWS 4096
#define C_COLS 10000
#define D_DIM  512
#define NT 5                      // 16-wide N tiles per wave (5*16 = 80 columns)
#define NCOLS (NT * 16)           // 80
#define LDS_ROW (D_DIM + 8)       // 520 bf16 = 1040 B row pitch -> lanes hit distinct banks
#define USE_ASYNC_LDS 1

__device__ __forceinline__ float sq8(const float4 a, const float4 b) {
  return a.x*a.x + a.y*a.y + a.z*a.z + a.w*a.w +
         b.x*b.x + b.y*b.y + b.z*b.z + b.w*b.w;
}

__device__ __forceinline__ void pack8(v16bf& v, int base, const float4 a, const float4 b) {
  v[base + 0] = (__bf16)a.x; v[base + 1] = (__bf16)a.y;
  v[base + 2] = (__bf16)a.z; v[base + 3] = (__bf16)a.w;
  v[base + 4] = (__bf16)b.x; v[base + 5] = (__bf16)b.y;
  v[base + 6] = (__bf16)b.z; v[base + 7] = (__bf16)b.w;
}

// ---------------------------------------------------------------------------
// Prep: fp32 -> bf16 copies of feat/weights + row norms f2[b], w2[c].
// One wave per row (512 f32 -> 16 f32/lane). 8 rows per 256-thread block.
// ---------------------------------------------------------------------------
__global__ __launch_bounds__(256)
void prep_kernel(const float* __restrict__ feat, const float* __restrict__ weights,
                 __bf16* __restrict__ feat_bf, __bf16* __restrict__ wt_bf,
                 float* __restrict__ f2, float* __restrict__ w2)
{
  const int lane = threadIdx.x & 31;
  const int wave = threadIdx.x >> 5;
  const int row  = blockIdx.x * 8 + wave;          // 0 .. B_ROWS+C_COLS-1

  const float* src; __bf16* dst; float* nrm;
  if (row < B_ROWS) {
    src = feat + (size_t)row * D_DIM;
    dst = feat_bf + (size_t)row * D_DIM;
    nrm = f2 + row;
  } else {
    const int r2 = row - B_ROWS;
    if (r2 >= C_COLS) return;
    src = weights + (size_t)r2 * D_DIM;
    dst = wt_bf + (size_t)r2 * D_DIM;
    nrm = w2 + r2;
  }

  const float4* sp = (const float4*)(src + lane * 16);
  float s = 0.0f;
  v16bf ov;
  const float4 x0 = sp[0], x1 = sp[1], x2 = sp[2], x3 = sp[3];
  s += sq8(x0, x1) + sq8(x2, x3);
  pack8(ov, 0, x0, x1);
  pack8(ov, 8, x2, x3);
  *(v8bf*)(dst + lane * 16)     = __builtin_shufflevector(ov, ov, 0,1,2,3,4,5,6,7);
  *(v8bf*)(dst + lane * 16 + 8) = __builtin_shufflevector(ov, ov, 8,9,10,11,12,13,14,15);

  s += __shfl_xor(s, 16, 32);
  s += __shfl_xor(s,  8, 32);
  s += __shfl_xor(s,  4, 32);
  s += __shfl_xor(s,  2, 32);
  s += __shfl_xor(s,  1, 32);
  if (lane == 0) *nrm = s;
}

// ---------------------------------------------------------------------------
// Fast GEMM: bf16 inputs, B block staged once per workgroup in LDS via
// global_load_async_to_lds, norms read from precomputed f2/w2.
// ---------------------------------------------------------------------------
__global__ __launch_bounds__(256)
void rbf_wmma_lds_kernel(const __bf16* __restrict__ feat_bf,
                         const __bf16* __restrict__ wt_bf,
                         const float* __restrict__ f2,
                         const float* __restrict__ w2,
                         float* __restrict__ out_train,
                         float* __restrict__ out_val)
{
  extern __shared__ __bf16 ldsB[];                 // NCOLS * LDS_ROW bf16 = 83,200 B
  const int tid  = threadIdx.x;
  const int lane = tid & 31;
  const int wave = tid >> 5;
  const int half = lane >> 4;                      // 0: lanes 0-15, 1: lanes 16-31
  const int lid  = lane & 15;

  const int m0 = (blockIdx.y * 8 + wave) * 16;     // 16 feat rows per wave
  const int n0 = blockIdx.x * NCOLS;               // 80 weight rows per block

  // ---- stage B block (80 rows x 512 bf16) into LDS asynchronously ----
  // 80 rows * 64 16B-segments = 5120 segments; 20 per thread, lanes contiguous.
  for (int c = tid; c < NCOLS * (D_DIM / 8); c += 256) {
    const int row = c >> 6;                        // 0..79
    const int seg = c & 63;                        // 16B segment in row
    const __bf16* g = wt_bf + (size_t)(n0 + row) * D_DIM + seg * 8;
    __bf16* l = &ldsB[row * LDS_ROW + seg * 8];
#if USE_ASYNC_LDS
    const unsigned lds_off = (unsigned)(size_t)l;  // low 32 bits of generic ptr = LDS byte addr
    asm volatile("global_load_async_to_lds_b128 %0, %1, off"
                 :: "v"(lds_off), "v"(g) : "memory");
#else
    *(v8bf*)l = *(const v8bf*)g;
#endif
  }
#if USE_ASYNC_LDS
  asm volatile("s_wait_asynccnt 0" ::: "memory");
#endif
  __syncthreads();

  // A (feat_bf): row m0+lid; per 32-chunk this lane-half owns K {16h..16h+15}
  const __bf16* ap = feat_bf + (size_t)(m0 + lid) * D_DIM + 16 * half;
  // B (LDS): col n0+t*16+lid; lane-half owns K {16h..16h+15}.
  // Two bases so every ds_load offset fits the 16-bit immediate field
  // (tile 4 starts at 66,560 B > 65,535).
  const __bf16* lb0 = &ldsB[lid * LDS_ROW + 16 * half];
  const __bf16* lb4 = lb0 + 4 * (16 * LDS_ROW);

  v8f acc[NT] = {};

  for (int kc = 0; kc < D_DIM; kc += 32) {
    const v8bf a_lo = *(const v8bf*)(ap + kc);
    const v8bf a_hi = *(const v8bf*)(ap + kc + 8);
    const v16bf av = __builtin_shufflevector(a_lo, a_hi,
                        0,1,2,3,4,5,6,7,8,9,10,11,12,13,14,15);

    // Issue all B-tile loads first (distinct registers) so the scheduler can
    // batch 10 ds_loads and retire WMMAs against a sliding dscnt, instead of
    // draining dscnt to 0 before every WMMA.
    v8bf blo[NT], bhi[NT];
#pragma unroll
    for (int t = 0; t < NT; ++t) {
      const __bf16* p = (t < 4 ? lb0 + t * (16 * LDS_ROW) : lb4) + kc;
      blo[t] = *(const v8bf*)(p);
      bhi[t] = *(const v8bf*)(p + 8);
    }
#pragma unroll
    for (int t = 0; t < NT; ++t) {
      const v16bf bv = __builtin_shufflevector(blo[t], bhi[t],
                          0,1,2,3,4,5,6,7,8,9,10,11,12,13,14,15);
      acc[t] = __builtin_amdgcn_wmma_f32_16x16x32_bf16(
          false, av, false, bv, (short)0, acc[t], false, false);
    }
  }

  float w2v[NT];
#pragma unroll
  for (int t = 0; t < NT; ++t) w2v[t] = w2[n0 + t * 16 + lid];

  // C/D layout: VGPR r, lanes 0-15 -> M=r, N=lane; lanes 16-31 -> M=r+8, N=lane-16.
#pragma unroll
  for (int r = 0; r < 8; ++r) {
    const int   mrow = r + 8 * half;
    const float f2v  = f2[m0 + mrow];
    const size_t rowoff = (size_t)(m0 + mrow) * C_COLS;
#pragma unroll
    for (int t = 0; t < NT; ++t) {
      float d = f2v + w2v[t] - 2.0f * acc[t][r];
      d = fmaxf(d, 0.0f);
      const float lg = __expf(-0.01f * d);
      const size_t idx = rowoff + (size_t)(n0 + t * 16 + lid);
      out_train[idx] = lg;
      out_val[idx]   = lg;
    }
  }
}

// ---------------------------------------------------------------------------
// Fallback (ws too small): self-contained fp32->bf16 in-loop version.
// ---------------------------------------------------------------------------
__global__ __launch_bounds__(256)
void rbf_wmma_kernel(const float* __restrict__ feat,
                     const float* __restrict__ weights,
                     float* __restrict__ out_train,
                     float* __restrict__ out_val)
{
  const int lane = threadIdx.x & 31;
  const int wave = threadIdx.x >> 5;
  const int half = lane >> 4;
  const int lid  = lane & 15;

  const int m0 = (blockIdx.y * 8 + wave) * 16;
  const int n0 = blockIdx.x * NCOLS;

  const float* ap = feat + (size_t)(m0 + lid) * D_DIM + 16 * half;
  const float* bp[NT];
#pragma unroll
  for (int t = 0; t < NT; ++t)
    bp[t] = weights + (size_t)(n0 + t * 16 + lid) * D_DIM + 16 * half;

  v8f acc[NT] = {};
  float fsq = 0.0f;
  float wsq[NT] = {};

  for (int kc = 0; kc < D_DIM; kc += 32) {
    const float4 a0 = *(const float4*)(ap + kc);
    const float4 a1 = *(const float4*)(ap + kc + 4);
    const float4 a2 = *(const float4*)(ap + kc + 8);
    const float4 a3 = *(const float4*)(ap + kc + 12);
    v16bf av;
    pack8(av, 0, a0, a1);
    pack8(av, 8, a2, a3);
    fsq += sq8(a0, a1) + sq8(a2, a3);

#pragma unroll
    for (int t = 0; t < NT; ++t) {
      const float4 w0 = *(const float4*)(bp[t] + kc);
      const float4 w1 = *(const float4*)(bp[t] + kc + 4);
      const float4 w2_ = *(const float4*)(bp[t] + kc + 8);
      const float4 w3 = *(const float4*)(bp[t] + kc + 12);
      v16bf bv;
      pack8(bv, 0, w0, w1);
      pack8(bv, 8, w2_, w3);
      wsq[t] += sq8(w0, w1) + sq8(w2_, w3);
      acc[t] = __builtin_amdgcn_wmma_f32_16x16x32_bf16(
          false, av, false, bv, (short)0, acc[t], false, false);
    }
  }

  fsq += __shfl_xor(fsq, 16, 32);
#pragma unroll
  for (int t = 0; t < NT; ++t) wsq[t] += __shfl_xor(wsq[t], 16, 32);

#pragma unroll
  for (int r = 0; r < 8; ++r) {
    const int   mrow = r + 8 * half;
    const float f2v  = __shfl(fsq, mrow, 32);
    const size_t rowoff = (size_t)(m0 + mrow) * C_COLS;
#pragma unroll
    for (int t = 0; t < NT; ++t) {
      float d = f2v + wsq[t] - 2.0f * acc[t][r];
      d = fmaxf(d, 0.0f);
      const float lg = __expf(-0.01f * d);
      const size_t idx = rowoff + (size_t)(n0 + t * 16 + lid);
      out_train[idx] = lg;
      out_val[idx]   = lg;
    }
  }
}

__global__ __launch_bounds__(256)
void copy_weights_kernel(const float4* __restrict__ src, float4* __restrict__ dst, int n4) {
  const int i = blockIdx.x * blockDim.x + threadIdx.x;
  if (i < n4) dst[i] = src[i];
}

extern "C" void kernel_launch(void* const* d_in, const int* in_sizes, int n_in,
                              void* d_out, int out_size, void* d_ws, size_t ws_size,
                              hipStream_t stream) {
  (void)in_sizes; (void)n_in; (void)out_size;
  const float* feat    = (const float*)d_in[0];
  const float* weights = (const float*)d_in[2];   // d_in[1] = int64 labels (unused)

  float* out       = (float*)d_out;
  float* out_train = out;
  float* out_val   = out + (size_t)B_ROWS * C_COLS;
  float* out_w     = out + 2 * (size_t)B_ROWS * C_COLS;

  // workspace layout (all section offsets are 256B-aligned)
  const size_t FEAT_BF_BYTES = (size_t)B_ROWS * D_DIM * 2;   //  4,194,304
  const size_t WT_BF_BYTES   = (size_t)C_COLS * D_DIM * 2;   // 10,240,000
  const size_t F2_BYTES      = (size_t)B_ROWS * 4;           //     16,384
  const size_t W2_BYTES      = (size_t)C_COLS * 4;           //     40,000
  const size_t WS_NEEDED = FEAT_BF_BYTES + WT_BF_BYTES + F2_BYTES + W2_BYTES;

  const dim3 grid(C_COLS / NCOLS, B_ROWS / (16 * 8));        // (125, 32), exact tiling
  const dim3 blk(256);

  if (ws_size >= WS_NEEDED) {
    char* ws = (char*)d_ws;
    __bf16* feat_bf = (__bf16*)(ws);
    __bf16* wt_bf   = (__bf16*)(ws + FEAT_BF_BYTES);
    float*  f2      = (float*)(ws + FEAT_BF_BYTES + WT_BF_BYTES);
    float*  w2      = (float*)(ws + FEAT_BF_BYTES + WT_BF_BYTES + F2_BYTES);

    const int nrows = B_ROWS + C_COLS;                       // 14096
    prep_kernel<<<dim3((nrows + 7) / 8), blk, 0, stream>>>(
        feat, weights, feat_bf, wt_bf, f2, w2);

    const size_t lds_bytes = (size_t)NCOLS * LDS_ROW * sizeof(__bf16);  // 83,200
    rbf_wmma_lds_kernel<<<grid, blk, lds_bytes, stream>>>(
        feat_bf, wt_bf, f2, w2, out_train, out_val);
  } else {
    rbf_wmma_kernel<<<grid, blk, 0, stream>>>(feat, weights, out_train, out_val);
  }

  const int n4 = (C_COLS * D_DIM) / 4;                       // 1,280,000 float4s
  copy_weights_kernel<<<dim3((n4 + 255) / 256), blk, 0, stream>>>(
      (const float4*)weights, (float4*)out_w, n4);
}
